// ScaleDotProuctAttention_19481971654760
// MI455X (gfx1250) — matrix operand order repaired and checked by reading, not compiled, not run on hardware
//
#include <hip/hip_runtime.h>
#include <math.h>

typedef __attribute__((ext_vector_type(16))) _Float16 v16h;
typedef __attribute__((ext_vector_type(8)))  float    v8f;

#define B_ 2
#define H_ 16
#define L_ 2048
#define D_ 64

// LDS swizzle: logical (m,n) within a 16x64 head tile -> physical float index.
// Additive swizzle keeps every access phase conflict-free with zero padding
// (row stride stays 64 words, so 16 heads * 16 * 64 * 4B = exactly 64 KB).
#define SWZ(m, n) (((m) << 6) | (((n) + 4 * (m)) & 63))

// ---------------------------------------------------------------------------
// Fully fused attention kernel.
// One block = 512 threads = 16 waves owns (b, 16-row m-tile) for ALL heads
// (the softmax couples heads, so all 16 score tiles must coexist).
// Loop over 32 n-strips of width 64:
//   1) wave h: scores S = (Q @ Kbuf)/8 via 8x v_wmma_f32_16x16x32_f16 -> LDS
//   2) block: softmax over the HEAD axis (reference softmax(axis=1)),
//      stream normalized weights to HBM (non-temporal; write-only data),
//      write normalized values back to LDS
//   3) wave h: O_acc += W_tile @ V_strip (8 more WMMAs), accumulators live
//      in registers for the entire n loop -> no 537 MB weight re-read.
// ---------------------------------------------------------------------------
__global__ __launch_bounds__(512)
void attn_fused(const float* __restrict__ Q,
                const float* __restrict__ K,
                const float* __restrict__ V,
                float* __restrict__ Wt,
                float* __restrict__ O) {
  const int tid  = threadIdx.x;
  const int lane = tid & 31;
  const int h    = tid >> 5;      // wave id == head id, 0..15
  const int half = lane >> 4;     // lane group 0/1
  const int l16  = lane & 15;

  const int m0 = blockIdx.x * 16; // 128 m-tiles
  const int b  = blockIdx.y;      // 2

  __shared__ float sTile[H_][16 * 64];   // 64 KB, swizzled via SWZ

  const size_t headOff = ((size_t)b * H_ + h) * (size_t)L_ * D_;
  const float* q  = Q + headOff;   // [L][D] row-major
  const float* kb = K + headOff;   // reshape quirk: key buffer viewed as [D][L]
  const float* vp = V + headOff;   // [L][D] row-major
  float*       op = O + headOff;

  const size_t hstride = (size_t)L_ * L_;
  float* wb = Wt + (size_t)b * H_ * hstride;

  // ---- persistent A fragments of Q (rows m0..m0+15, K-dim 0..63) ----
  // 16-bit A 16x32 layout (contiguous, matches B): lanes 0-15: row l16, K 0..15;
  // lanes 16-31: row l16, K 16..31 (per 32-wide k-step).
  v16h aq[2];
  {
    const float* qrow = q + (size_t)(m0 + l16) * D_;
    #pragma unroll
    for (int ks = 0; ks < 2; ++ks) {
      const int kbase = ks * 32 + half * 16;
      #pragma unroll
      for (int i = 0; i < 8; ++i) {
        aq[ks][i]     = (_Float16)qrow[kbase + i];
        aq[ks][i + 8] = (_Float16)qrow[kbase + 8 + i];
      }
    }
  }

  // ---- persistent output accumulators: 16 rows x 64 cols in f32 ----
  v8f co[4];
  #pragma unroll
  for (int nt = 0; nt < 4; ++nt) co[nt] = (v8f)0.f;

  for (int n0 = 0; n0 < L_; n0 += 64) {
    // ================= 1) scores for this head's 16x64 tile =================
    v8f cs[4];
    #pragma unroll
    for (int nt = 0; nt < 4; ++nt) cs[nt] = (v8f)0.f;

    #pragma unroll
    for (int ks = 0; ks < 2; ++ks) {
      #pragma unroll
      for (int nt = 0; nt < 4; ++nt) {
        // B 32x16: lane column n = l16; lanes 0-15 K 0..15, lanes 16-31 K 16..31
        v16h bf;
        const float* col = kb + (size_t)(ks * 32 + half * 16) * L_
                              + (size_t)(n0 + nt * 16 + l16);
        #pragma unroll
        for (int i = 0; i < 16; ++i) bf[i] = (_Float16)col[(size_t)i * L_];
        cs[nt] = __builtin_amdgcn_wmma_f32_16x16x32_f16(
            false, aq[ks], false, bf, (short)0, cs[nt], false, false);
      }
    }

    // stash scaled scores: C/D layout -> vgpr j = row 8*half + j, col l16
    #pragma unroll
    for (int nt = 0; nt < 4; ++nt) {
      #pragma unroll
      for (int j = 0; j < 8; ++j) {
        sTile[h][SWZ(half * 8 + j, nt * 16 + l16)] = cs[nt][j] * 0.125f;
      }
    }
    __syncthreads();

    // ================= 2) softmax over heads + weight streaming =============
    #pragma unroll
    for (int rep = 0; rep < 2; ++rep) {
      const int p  = tid + rep * 512;    // 0..1023 tile positions
      const int ml = p >> 6;
      const int nl = p & 63;
      const int sp = SWZ(ml, nl);
      float v[H_];
      float mx = -3.402823466e38f;
      #pragma unroll
      for (int hh = 0; hh < H_; ++hh) {
        v[hh] = sTile[hh][sp];
        mx = fmaxf(mx, v[hh]);
      }
      float sum = 0.f;
      #pragma unroll
      for (int hh = 0; hh < H_; ++hh) {
        v[hh] = __expf(v[hh] - mx);
        sum += v[hh];
      }
      const float inv = 1.0f / sum;
      float* dst = wb + (size_t)(m0 + ml) * L_ + (size_t)(n0 + nl);
      #pragma unroll
      for (int hh = 0; hh < H_; ++hh) {
        const float wv = v[hh] * inv;
        __builtin_nontemporal_store(wv, dst + (size_t)hh * hstride);
        sTile[hh][sp] = wv;              // feed the second GEMM from LDS
      }
    }
    __syncthreads();

    // ================= 3) O_acc += W_tile @ V_strip =========================
    // A2 = normalized weights, 16(m) x 64(n-reduction), from swizzled LDS.
    v16h a2[2];
    #pragma unroll
    for (int ks = 0; ks < 2; ++ks) {
      const int kbase = ks * 32 + half * 16;
      #pragma unroll
      for (int i = 0; i < 8; ++i) {
        a2[ks][i]     = (_Float16)sTile[h][SWZ(l16, kbase + i)];
        a2[ks][i + 8] = (_Float16)sTile[h][SWZ(l16, kbase + 8 + i)];
      }
    }
    #pragma unroll
    for (int ks = 0; ks < 2; ++ks) {
      #pragma unroll
      for (int nt = 0; nt < 4; ++nt) {
        // B2 32x16 from V rows n0+.. (L2-resident), columns d
        v16h bf;
        const float* col = vp + (size_t)(n0 + ks * 32 + half * 16) * D_
                              + (size_t)(nt * 16 + l16);
        #pragma unroll
        for (int i = 0; i < 16; ++i) bf[i] = (_Float16)col[i * D_];
        co[nt] = __builtin_amdgcn_wmma_f32_16x16x32_f16(
            false, a2[ks], false, bf, (short)0, co[nt], false, false);
      }
    }
    __syncthreads();   // protect sTile before next strip's stash
  }

  // ---- write the finished output tile (16 x 64) ----
  #pragma unroll
  for (int nt = 0; nt < 4; ++nt) {
    float* orow = op + (size_t)(m0 + half * 8) * D_ + (size_t)(nt * 16 + l16);
    #pragma unroll
    for (int j = 0; j < 8; ++j) orow[(size_t)j * D_] = co[nt][j];
  }
}

extern "C" void kernel_launch(void* const* d_in, const int* in_sizes, int n_in,
                              void* d_out, int out_size, void* d_ws, size_t ws_size,
                              hipStream_t stream) {
  (void)in_sizes; (void)n_in; (void)out_size; (void)d_ws; (void)ws_size;
  const float* Q = (const float*)d_in[0];
  const float* K = (const float*)d_in[1];
  const float* V = (const float*)d_in[2];

  float* out = (float*)d_out;
  float* O   = out;                                   // [B,H,L,D] first
  float* Wt  = out + (size_t)B_ * H_ * L_ * D_;       // then weights [B,H,L,L]

  dim3 grid(L_ / 16, B_);             // (128, 2) blocks, 512 threads = 16 waves
  attn_fused<<<grid, 512, 0, stream>>>(Q, K, V, Wt, O);
}
